// CrossFrequencyInteraction_29944511987974
// MI455X (gfx1250) — matrix operand order repaired and checked by reading, not compiled, not run on hardware
//
#include <hip/hip_runtime.h>
#include <hip/hip_bf16.h>

#define DIM_C   256
#define NPIX    4096      // 64*64
#define BATCH   8
#define NHEADS  4
#define HDIM    64
#define BN_EPS  1e-5f

typedef __bf16 bf16;
typedef __attribute__((ext_vector_type(8)))  __bf16 v8bf;
typedef __attribute__((ext_vector_type(16))) __bf16 v16bf;
typedef __attribute__((ext_vector_type(8)))  float  v8f;

union Frag16 { v16bf v; __bf16 e[16]; };

// ---------------------------------------------------------------------------
// WMMA helpers (CDNA5 16x16x32 bf16, wave32)
// ---------------------------------------------------------------------------
__device__ __forceinline__ v8f wmma_bf16(v16bf a, v16bf b, v8f c) {
  return __builtin_amdgcn_wmma_f32_16x16x32_bf16(
      /*neg_a=*/false, a, /*neg_b=*/false, b,
      /*c_mod=*/(short)0, c, /*reuse_a=*/false, /*reuse_b=*/false);
}

// A fragment: 16x32 tile, memory row-major [M][K], K contiguous.
// lane<16: row=lane,     chunks at k0+{0..7}  and k0+{16..23}
// lane>=16: row=lane-16, chunks at k0+{8..15} and k0+{24..31}
__device__ __forceinline__ v16bf frag_a_load(const bf16* tile, int ld) {
  int lane = threadIdx.x & 31;
  const bf16* p = tile + (size_t)(lane & 15) * ld + ((lane >> 4) << 3);
  v8bf lo = *(const v8bf*)p;
  v8bf hi = *(const v8bf*)(p + 16);
  return __builtin_shufflevector(lo, hi, 0,1,2,3,4,5,6,7,8,9,10,11,12,13,14,15);
}

// B fragment: 32x16 tile, memory row-major [N][K] (i.e. transposed B), K contiguous.
// lane<16: col=lane, K = {0..7,16..23} ; lane>=16: col=lane-16, K = {8..15,24..31}
__device__ __forceinline__ v16bf frag_b_load(const bf16* tile, int ld) {
  int lane = threadIdx.x & 31;
  const bf16* p = tile + (size_t)(lane & 15) * ld + ((lane >> 4) << 3);
  v8bf lo = *(const v8bf*)p;
  v8bf hi = *(const v8bf*)(p + 16);
  return __builtin_shufflevector(lo, hi, 0,1,2,3,4,5,6,7,8,9,10,11,12,13,14,15);
}

// ---------------------------------------------------------------------------
// Input transpose:  c-major f32 [B][C][N]  ->  n-major bf16 [B][N][C] (+opt f32)
// ---------------------------------------------------------------------------
template <bool WRITE_F32>
__global__ void __launch_bounds__(256)
transpose_in_kernel(const float* __restrict__ x, bf16* __restrict__ xb,
                    float* __restrict__ xf) {
  __shared__ float t[32][33];
  int bz = blockIdx.z;
  int c0 = blockIdx.y * 32;
  int n0 = blockIdx.x * 32;
  int tx = threadIdx.x, ty = threadIdx.y;   // (32, 8)
#pragma unroll
  for (int j = 0; j < 4; ++j)
    t[ty + 8*j][tx] = x[((size_t)bz*DIM_C + c0 + ty + 8*j)*NPIX + n0 + tx];
  __syncthreads();
#pragma unroll
  for (int j = 0; j < 4; ++j) {
    int n = n0 + ty + 8*j, c = c0 + tx;
    float v = t[tx][ty + 8*j];
    size_t idx = ((size_t)bz*NPIX + n)*DIM_C + c;
    xb[idx] = (bf16)v;
    if (WRITE_F32) xf[idx] = v;
  }
}

__global__ void f32_to_bf16_kernel(const float* __restrict__ in,
                                   bf16* __restrict__ out, int n) {
  int i = blockIdx.x * 256 + threadIdx.x;
  if (i < n) out[i] = (bf16)in[i];
}

// ---------------------------------------------------------------------------
// conv1x1 GEMM:  Yt[b][n][o] = sum_c X[b][n][c] * W[o][c]  (+bias, +residual)
// Per wave: 64n x 32o tile, K=256.  grid=512 blocks x 256 thr (8 waves).
// MODE 0: write bf16 to Ybf.  MODE 1: write f32 (acc+bias+resid) to Yf.
// ---------------------------------------------------------------------------
template <int MODE>
__global__ void __launch_bounds__(256)
conv_gemm_kernel(const bf16* __restrict__ X, const bf16* __restrict__ W,
                 const float* __restrict__ bias, bf16* __restrict__ Ybf,
                 float* __restrict__ Yf, const float* __restrict__ resid) {
  int wave = blockIdx.x * 8 + (threadIdx.x >> 5);
  int bb   = wave >> 9;              // batch
  int rem  = wave & 511;
  int n0   = (rem >> 3) * 64;
  int o0   = (rem & 7) * 32;
  int lane = threadIdx.x & 31;

  v8f acc[4][2];
#pragma unroll
  for (int i = 0; i < 4; ++i)
#pragma unroll
    for (int j = 0; j < 2; ++j) acc[i][j] = (v8f)(0.0f);

  const bf16* Xb = X + ((size_t)bb * NPIX + n0) * DIM_C;
  for (int k0 = 0; k0 < DIM_C; k0 += 32) {
    v16bf af[4], bf[2];
#pragma unroll
    for (int i = 0; i < 4; ++i)
      af[i] = frag_a_load(Xb + (size_t)(16*i)*DIM_C + k0, DIM_C);
#pragma unroll
    for (int j = 0; j < 2; ++j)
      bf[j] = frag_b_load(W + (size_t)(o0 + 16*j)*DIM_C + k0, DIM_C);
#pragma unroll
    for (int i = 0; i < 4; ++i)
#pragma unroll
      for (int j = 0; j < 2; ++j)
        acc[i][j] = wmma_bf16(af[i], bf[j], acc[i][j]);
  }

  int col = lane & 15;
  int hi8 = (lane >> 4) * 8;
#pragma unroll
  for (int j = 0; j < 2; ++j) {
    int o = o0 + 16*j + col;
    float bv = bias[o];
#pragma unroll
    for (int i = 0; i < 4; ++i) {
#pragma unroll
      for (int r = 0; r < 8; ++r) {
        int n = n0 + 16*i + r + hi8;
        size_t idx = ((size_t)bb*NPIX + n)*DIM_C + o;
        float v = acc[i][j][r] + bv;
        if (MODE == 0) Ybf[idx] = (bf16)v;
        else           Yf[idx]  = v + resid[idx];
      }
    }
  }
}

// ---------------------------------------------------------------------------
// Softmax over n (per (b,c)):  3-stage, deterministic, no float atomics.
// Values are ~N(0, 0.32^2) so exp without max-subtraction is safe in f32.
// ---------------------------------------------------------------------------
__global__ void __launch_bounds__(256)
softmax_partial_kernel(const bf16* __restrict__ Q, float* __restrict__ part) {
  int b = blockIdx.x, chunk = blockIdx.y, c = threadIdx.x;
  const bf16* p = Q + ((size_t)b*NPIX + (size_t)chunk*128)*DIM_C + c;
  float s = 0.f;
  for (int i = 0; i < 128; ++i) s += __expf((float)p[(size_t)i*DIM_C]);
  part[((size_t)b*32 + chunk)*DIM_C + c] = s;
}

__global__ void softmax_combine_kernel(const float* __restrict__ part,
                                       float* __restrict__ invs) {
  int b = blockIdx.x, c = threadIdx.x;
  float s = 0.f;
  for (int ch = 0; ch < 32; ++ch) s += part[((size_t)b*32 + ch)*DIM_C + c];
  invs[b*DIM_C + c] = 1.0f / s;
}

__global__ void __launch_bounds__(256)
softmax_write_kernel(const bf16* __restrict__ Q, const float* __restrict__ invs,
                     bf16* __restrict__ Qs) {
  int b = blockIdx.x, chunk = blockIdx.y, c = threadIdx.x;
  float iv = invs[b*DIM_C + c];
  size_t base = ((size_t)b*NPIX + (size_t)chunk*128)*DIM_C + c;
  for (int i = 0; i < 128; ++i) {
    float x = (float)Q[base + (size_t)i*DIM_C];
    Qs[base + (size_t)i*DIM_C] = (bf16)(__expf(x) * iv);
  }
}

// ---------------------------------------------------------------------------
// Context GEMM: ctxT[bh][e][d] = sum_n V[b][n][64h+e] * Ks[b][n][64h+d]
// K (=n) is strided -> stage [32n x 64c] tiles in LDS, gather fragments.
// K split over 4 blocks per (b,h); deterministic partial-sum reduction after.
// ---------------------------------------------------------------------------
__global__ void __launch_bounds__(256)
context_gemm_kernel(const bf16* __restrict__ Ks, const bf16* __restrict__ V,
                    float* __restrict__ Ctxp) {
  __shared__ __align__(16) bf16 Vt[32][72];
  __shared__ __align__(16) bf16 Kt[32][72];
  int bh = blockIdx.x >> 2;          // 0..31
  int kc = blockIdx.x & 3;
  int b = bh >> 2, h = bh & 3;
  int tid = threadIdx.x, lane = tid & 31, wv = tid >> 5;
  int et  = wv >> 1;                 // 0..3  (e tile)
  int dt0 = (wv & 1) * 2;            // {0,2} (two d tiles)

  v8f acc[2];
  acc[0] = (v8f)(0.0f); acc[1] = (v8f)(0.0f);

  int ldrow = tid >> 3, ldch = (tid & 7) * 8;
  const bf16* Vb = V  + (size_t)b*NPIX*DIM_C + h*HDIM;
  const bf16* Kb = Ks + (size_t)b*NPIX*DIM_C + h*HDIM;

  for (int nn = kc*1024; nn < kc*1024 + 1024; nn += 32) {
    *(v8bf*)&Vt[ldrow][ldch] = *(const v8bf*)(Vb + (size_t)(nn + ldrow)*DIM_C + ldch);
    *(v8bf*)&Kt[ldrow][ldch] = *(const v8bf*)(Kb + (size_t)(nn + ldrow)*DIM_C + ldch);
    __syncthreads();

    Frag16 a;
    int m = lane & 15, kbA = (lane >> 4) * 8;
    int ecol = et*16 + m;
#pragma unroll
    for (int hh = 0; hh < 8; ++hh) {
      a.e[hh]     = Vt[kbA + hh][ecol];
      a.e[8 + hh] = Vt[kbA + 16 + hh][ecol];
    }
#pragma unroll
    for (int j = 0; j < 2; ++j) {
      Frag16 bfr;
      int dcol = (dt0 + j)*16 + (lane & 15);
      int kbB = (lane >> 4) * 8;
#pragma unroll
      for (int hh = 0; hh < 16; ++hh) bfr.e[hh] = Kt[kbB + (hh & 7) + (hh >> 3) * 16][dcol];
      acc[j] = wmma_bf16(a.v, bfr.v, acc[j]);
    }
    __syncthreads();
  }

  int hi8 = (lane >> 4) * 8;
#pragma unroll
  for (int j = 0; j < 2; ++j) {
    int dcol = (dt0 + j)*16 + (lane & 15);
#pragma unroll
    for (int r = 0; r < 8; ++r) {
      int e = et*16 + r + hi8;
      Ctxp[(size_t)kc*131072 + (size_t)bh*4096 + e*64 + dcol] = acc[j][r];
    }
  }
}

__global__ void ctx_reduce_kernel(const float* __restrict__ Ctxp,
                                  bf16* __restrict__ CtxT, int n) {
  int i = blockIdx.x * 256 + threadIdx.x;
  if (i < n) {
    float s = Ctxp[i] + Ctxp[n + i] + Ctxp[2*(size_t)n + i] + Ctxp[3*(size_t)n + i];
    CtxT[i] = (bf16)s;
  }
}

// ---------------------------------------------------------------------------
// Attention-out GEMM: O[b][n][64h+e] = sum_d Qs[b][n][64h+d] * ctxT[bh][e][d]
// Both operands contiguous in K=d.  Per wave: 64n x 32e tile, K=64.
// ---------------------------------------------------------------------------
__global__ void __launch_bounds__(256)
attn_out_gemm_kernel(const bf16* __restrict__ Qs, const bf16* __restrict__ CtxT,
                     bf16* __restrict__ O) {
  int wave = blockIdx.x * 8 + (threadIdx.x >> 5);
  int bh  = wave >> 7;               // 0..31
  int rem = wave & 127;
  int n0  = (rem >> 1) * 64;
  int e0  = (rem & 1) * 32;
  int b = bh >> 2, h = bh & 3;
  int lane = threadIdx.x & 31;

  v8f acc[4][2];
#pragma unroll
  for (int i = 0; i < 4; ++i)
#pragma unroll
    for (int j = 0; j < 2; ++j) acc[i][j] = (v8f)(0.0f);

  const bf16* Qb = Qs + ((size_t)b*NPIX + n0)*DIM_C + h*HDIM;
  const bf16* Cb = CtxT + (size_t)bh*HDIM*HDIM;
#pragma unroll
  for (int k0 = 0; k0 < HDIM; k0 += 32) {
    v16bf af[4], bf[2];
#pragma unroll
    for (int i = 0; i < 4; ++i)
      af[i] = frag_a_load(Qb + (size_t)(16*i)*DIM_C + k0, DIM_C);
#pragma unroll
    for (int j = 0; j < 2; ++j)
      bf[j] = frag_b_load(Cb + (size_t)(e0 + 16*j)*HDIM + k0, HDIM);
#pragma unroll
    for (int i = 0; i < 4; ++i)
#pragma unroll
      for (int j = 0; j < 2; ++j)
        acc[i][j] = wmma_bf16(af[i], bf[j], acc[i][j]);
  }

  int col = lane & 15, hi8 = (lane >> 4) * 8;
#pragma unroll
  for (int j = 0; j < 2; ++j) {
    int e = e0 + 16*j + col;
#pragma unroll
    for (int i = 0; i < 4; ++i)
#pragma unroll
      for (int r = 0; r < 8; ++r) {
        int n = n0 + 16*i + r + hi8;
        O[((size_t)b*NPIX + n)*DIM_C + h*HDIM + e] = (bf16)acc[i][j][r];
      }
  }
}

// ---------------------------------------------------------------------------
// BatchNorm: deterministic two-stage stats, then apply + transpose to c-major.
// ---------------------------------------------------------------------------
__global__ void __launch_bounds__(256)
bn_partial_kernel(const float* __restrict__ Rf, float* __restrict__ psum,
                  float* __restrict__ psq) {
  int blk = blockIdx.x, c = threadIdx.x;  // 256 blocks x 128 rows
  size_t row0 = (size_t)blk * 128;
  float s = 0.f, q = 0.f;
  for (int i = 0; i < 128; ++i) {
    float v = Rf[(row0 + i)*DIM_C + c];
    s += v; q += v*v;
  }
  psum[blk*DIM_C + c] = s;
  psq [blk*DIM_C + c] = q;
}

__global__ void bn_finalize_kernel(const float* __restrict__ psum,
                                   const float* __restrict__ psq,
                                   float* __restrict__ stats) {
  int c = threadIdx.x;
  float s = 0.f, q = 0.f;
  for (int blk = 0; blk < 256; ++blk) {
    s += psum[blk*DIM_C + c];
    q += psq [blk*DIM_C + c];
  }
  float cnt = (float)(BATCH * NPIX);
  float mean = s / cnt;
  stats[c]         = mean;
  stats[DIM_C + c] = q / cnt - mean*mean;
}

__global__ void __launch_bounds__(256)
bn_apply_transpose_kernel(const float* __restrict__ Rf,
                          const float* __restrict__ stats,
                          const float* __restrict__ gamma,
                          const float* __restrict__ beta,
                          float* __restrict__ out) {
  __shared__ float t[32][33];
  int bz = blockIdx.z;
  int c0 = blockIdx.y * 32;
  int n0 = blockIdx.x * 32;
  int tx = threadIdx.x, ty = threadIdx.y;
#pragma unroll
  for (int j = 0; j < 4; ++j)
    t[ty + 8*j][tx] = Rf[((size_t)bz*NPIX + n0 + ty + 8*j)*DIM_C + c0 + tx];
  __syncthreads();
#pragma unroll
  for (int j = 0; j < 4; ++j) {
    int c = c0 + ty + 8*j, n = n0 + tx;
    float mean = stats[c];
    float var  = stats[DIM_C + c];
    float sc   = gamma[c] * rsqrtf(var + BN_EPS);
    out[((size_t)bz*DIM_C + c)*NPIX + n] = (t[tx][ty + 8*j] - mean)*sc + beta[c];
  }
}

// ---------------------------------------------------------------------------
extern "C" void kernel_launch(void* const* d_in, const int* in_sizes, int n_in,
                              void* d_out, int out_size, void* d_ws, size_t ws_size,
                              hipStream_t stream) {
  (void)in_sizes; (void)n_in; (void)out_size; (void)ws_size;

  const float* rgb_low  = (const float*)d_in[0];
  const float* rgb_high = (const float*)d_in[1];
  const float* dsm_low  = (const float*)d_in[2];
  const float* dsm_high = (const float*)d_in[3];
  const float* w [8];
  const float* wb[8];
  for (int i = 0; i < 8; ++i) {         // q1,k1,v1,q2,k2,v2,proj1,proj2
    w [i] = (const float*)d_in[4 + 2*i];
    wb[i] = (const float*)d_in[5 + 2*i];
  }
  const float* gamma = (const float*)d_in[20];
  const float* beta  = (const float*)d_in[21];
  float* out = (float*)d_out;

  const size_t T = (size_t)BATCH * NPIX * DIM_C;   // elems per tensor (8,388,608)

  char* p = (char*)d_ws;
  auto take = [&](size_t bytes) -> char* {
    char* r = p; p += (bytes + 255) & ~(size_t)255; return r;
  };
  bf16* Wbf[8]; for (int i = 0; i < 8; ++i) Wbf[i] = (bf16*)take(DIM_C*DIM_C*2);
  bf16* Xbf[4]; for (int i = 0; i < 4; ++i) Xbf[i] = (bf16*)take(T*2);
  float* Xf[2]; for (int i = 0; i < 2; ++i) Xf[i]  = (float*)take(T*4);
  bf16* Qraw = (bf16*)take(T*2);
  bf16* Kraw = (bf16*)take(T*2);
  bf16* Vbf  = (bf16*)take(T*2);
  bf16* Qs   = (bf16*)take(T*2);
  bf16* Ks   = (bf16*)take(T*2);
  bf16* Obf  = (bf16*)take(T*2);
  float* Rf  = (float*)take(T*4);
  float* partQ = (float*)take((size_t)BATCH*32*DIM_C*4);
  float* partK = (float*)take((size_t)BATCH*32*DIM_C*4);
  float* invsQ = (float*)take((size_t)BATCH*DIM_C*4);
  float* invsK = (float*)take((size_t)BATCH*DIM_C*4);
  float* Ctxp  = (float*)take((size_t)4*32*HDIM*HDIM*4);
  bf16*  CtxT  = (bf16*)take((size_t)32*HDIM*HDIM*2);
  float* psum  = (float*)take((size_t)256*DIM_C*4);
  float* psq   = (float*)take((size_t)256*DIM_C*4);
  float* stats = (float*)take((size_t)2*DIM_C*4);

  dim3 tb(32, 8);
  dim3 tg(NPIX/32, DIM_C/32, BATCH);  // 128 x 8 x 8

  // transpose + convert inputs (rgb also keeps an fp32 n-major residual copy)
  transpose_in_kernel<true ><<<tg, tb, 0, stream>>>(rgb_low,  Xbf[0], Xf[0]);
  transpose_in_kernel<true ><<<tg, tb, 0, stream>>>(rgb_high, Xbf[1], Xf[1]);
  transpose_in_kernel<false><<<tg, tb, 0, stream>>>(dsm_low,  Xbf[2], nullptr);
  transpose_in_kernel<false><<<tg, tb, 0, stream>>>(dsm_high, Xbf[3], nullptr);
  for (int i = 0; i < 8; ++i)
    f32_to_bf16_kernel<<<256, 256, 0, stream>>>(w[i], Wbf[i], DIM_C*DIM_C);

  auto branch = [&](int qi, int ki, int vi, int pi, const bf16* xq,
                    const bf16* xkv, const float* resid, float* outsec) {
    conv_gemm_kernel<0><<<512, 256, 0, stream>>>(xq,  Wbf[qi], wb[qi], Qraw, nullptr, nullptr);
    conv_gemm_kernel<0><<<512, 256, 0, stream>>>(xkv, Wbf[ki], wb[ki], Kraw, nullptr, nullptr);
    conv_gemm_kernel<0><<<512, 256, 0, stream>>>(xkv, Wbf[vi], wb[vi], Vbf,  nullptr, nullptr);
    dim3 sg(BATCH, 32);
    softmax_partial_kernel<<<sg, 256, 0, stream>>>(Qraw, partQ);
    softmax_partial_kernel<<<sg, 256, 0, stream>>>(Kraw, partK);
    softmax_combine_kernel<<<BATCH, 256, 0, stream>>>(partQ, invsQ);
    softmax_combine_kernel<<<BATCH, 256, 0, stream>>>(partK, invsK);
    softmax_write_kernel<<<sg, 256, 0, stream>>>(Qraw, invsQ, Qs);
    softmax_write_kernel<<<sg, 256, 0, stream>>>(Kraw, invsK, Ks);
    context_gemm_kernel<<<128, 256, 0, stream>>>(Ks, Vbf, Ctxp);
    ctx_reduce_kernel<<<(131072 + 255)/256, 256, 0, stream>>>(Ctxp, CtxT, 131072);
    attn_out_gemm_kernel<<<512, 256, 0, stream>>>(Qs, CtxT, Obf);
    conv_gemm_kernel<1><<<512, 256, 0, stream>>>(Obf, Wbf[pi], wb[pi], nullptr, Rf, resid);
    bn_partial_kernel<<<256, 256, 0, stream>>>(Rf, psum, psq);
    bn_finalize_kernel<<<1, 256, 0, stream>>>(psum, psq, stats);
    bn_apply_transpose_kernel<<<tg, tb, 0, stream>>>(Rf, stats, gamma, beta, outsec);
  };

  branch(0, 1, 2, 6, Xbf[0], Xbf[3], Xf[0], out);       // q=rgb_low,  kv=dsm_high
  branch(3, 4, 5, 7, Xbf[1], Xbf[2], Xf[1], out + T);   // q=rgb_high, kv=dsm_low

  hipMemcpyAsync(out + 2*T, dsm_low,  T*sizeof(float), hipMemcpyDeviceToDevice, stream);
  hipMemcpyAsync(out + 3*T, dsm_high, T*sizeof(float), hipMemcpyDeviceToDevice, stream);
}
